// NanoLlamaAttention_62294205662116
// MI455X (gfx1250) — matrix operand order repaired and checked by reading, not compiled, not run on hardware
//
#include <hip/hip_runtime.h>
#include <hip/hip_bf16.h>

// ---------------------------------------------------------------------------
// NanoLlamaAttention forward for MI455X (gfx1250), bf16 WMMA everywhere.
// B=2, S=2048, H=2048, NH=16, HD=128. ~206 GFLOP, L2-resident -> compute bound.
// ---------------------------------------------------------------------------

typedef __attribute__((ext_vector_type(16))) __bf16 v16bf;
typedef __attribute__((ext_vector_type(8)))  __bf16 v8bf;
typedef __attribute__((ext_vector_type(8)))  float  v8f;

#define DEVI static __device__ __forceinline__

constexpr int BATCH = 2;
constexpr int SEQ   = 2048;
constexpr int HDIM  = 2048;
constexpr int NHEADS = 16;
constexpr int HSZ   = 128;               // head dim
constexpr int MTOT  = BATCH * SEQ;       // 4096
constexpr float SM_SCALE = 0.08838834764831843f; // 1/sqrt(128)

union V16U { v16bf v; v8bf h[2]; };

// A fragment (16x32 bf16, M x K): lane l -> row m0 + (l&15); K elements
//   k0 + 16*(l>>4) + 0..15 (contiguous, matches B fragment)
DEVI v16bf load_a_frag(const __bf16* base, int ld, int m0, int k0, int l15, int half) {
  const __bf16* p = base + (size_t)(m0 + l15) * ld + k0 + half * 16;
  V16U u;
  u.h[0] = *(const v8bf*)p;
  u.h[1] = *(const v8bf*)(p + 8);
  return u.v;
}

// B fragment (32x16 bf16, K x N) from a row-major [N][K] source:
//   lane l -> column n0 + (l&15), K elements k0 + 16*(l>>4) + 0..15 (contiguous)
DEVI v16bf load_b_frag(const __bf16* base, int ld, int n0, int k0, int l15, int half) {
  const __bf16* p = base + (size_t)(n0 + l15) * ld + k0 + half * 16;
  return *(const v16bf*)p;
}

DEVI v8f wmma_bf16(v16bf a, v16bf b, v8f c) {
  // (neg_a, A, neg_b, B, c_mod, C, reuse_a, reuse_b)
  return __builtin_amdgcn_wmma_f32_16x16x32_bf16(false, a, false, b, (short)0, c,
                                                 false, false);
}

// reductions across the 16 lanes of a half-wave (row of a D fragment)
DEVI float red16_max(float v) {
  v = fmaxf(v, __shfl_xor(v, 1, 32));
  v = fmaxf(v, __shfl_xor(v, 2, 32));
  v = fmaxf(v, __shfl_xor(v, 4, 32));
  v = fmaxf(v, __shfl_xor(v, 8, 32));
  return v;
}
DEVI float red16_sum(float v) {
  v += __shfl_xor(v, 1, 32);
  v += __shfl_xor(v, 2, 32);
  v += __shfl_xor(v, 4, 32);
  v += __shfl_xor(v, 8, 32);
  return v;
}

// ---------------------------------------------------------------------------
__global__ void cast_f32_bf16(const float* __restrict__ in,
                              __bf16* __restrict__ out, int n) {
  int i = blockIdx.x * blockDim.x + threadIdx.x;
  int stride = gridDim.x * blockDim.x;
  for (; i < n; i += stride) out[i] = (__bf16)in[i];
}

// ---------------------------------------------------------------------------
// GEMM: D = A[M][K] * W[N][K]^T. Each wave owns a 16x128 output tile
// (8 accumulator fragments -> each A fragment feeds 8 WMMAs).
// MODE_ROPE: apply RoPE, store bf16 to [B,NH,S,HSZ]
// MODE_V   : store bf16 transposed to [B,NH,HSZ,S]
// MODE_OUT : store fp32 to [M][H] (final result)
enum { MODE_ROPE = 0, MODE_V = 1, MODE_OUT = 2 };

template <int MODE>
__global__ __launch_bounds__(256, 1)
void gemm16x128(const __bf16* __restrict__ A,
                const __bf16* __restrict__ W,
                __bf16* __restrict__ outb,
                float* __restrict__ outf,
                const float* __restrict__ fcos,
                const float* __restrict__ fsin) {
  const int lane = threadIdx.x & 31;
  const int half = lane >> 4;
  const int l15  = lane & 15;
  const int wid  = blockIdx.x * (blockDim.x >> 5) + (threadIdx.x >> 5);
  const int m0   = (wid >> 4) << 4;   // HDIM/128 = 16 n-tiles per tile-row
  const int n0   = (wid & 15) << 7;

  v8f acc[8] = {};
  for (int k0 = 0; k0 < HDIM; k0 += 32) {
    if (k0 + 32 < HDIM) {  // global_prefetch_b8 of next A strip
      __builtin_prefetch(A + (size_t)(m0 + l15) * HDIM + k0 + 32, 0, 0);
    }
    v16bf a = load_a_frag(A, HDIM, m0, k0, l15, half);
#pragma unroll
    for (int f = 0; f < 8; ++f) {
      v16bf b = load_b_frag(W, HDIM, n0 + f * 16, k0, l15, half);
      acc[f] = wmma_bf16(a, b, acc[f]);
    }
  }

#pragma unroll
  for (int f = 0; f < 8; ++f) {
#pragma unroll
    for (int r = 0; r < 8; ++r) {
      const int m = m0 + r + half * 8;         // D row (ISA layout)
      const int n = n0 + f * 16 + l15;         // D col
      const int b = m / SEQ, s = m % SEQ;
      const int h = n / HSZ, d = n % HSZ;
      float v = acc[f][r];
      if constexpr (MODE == MODE_ROPE) {
        const float c  = fcos[s * (HSZ / 2) + (d >> 1)];
        const float sn = fsin[s * (HSZ / 2) + (d >> 1)];
        const float partner = __shfl_xor(v, 1, 32);  // neighbor column
        v = (d & 1) ? (partner * sn + v * c) : (v * c - partner * sn);
        outb[((size_t)(b * NHEADS + h) * SEQ + s) * HSZ + d] = (__bf16)v;
      } else if constexpr (MODE == MODE_V) {
        outb[((size_t)(b * NHEADS + h) * HSZ + d) * SEQ + s] = (__bf16)v;
      } else {
        outf[(size_t)m * HDIM + n] = v;
      }
    }
  }
}

// ---------------------------------------------------------------------------
// Flash attention: one wave per (b,h,16-row query tile). Streams 32 keys/step,
// online softmax, P goes through LDS to reach WMMA A layout, 8 WMMAs for P*V.
__global__ __launch_bounds__(256, 1)
void flash_attn(const __bf16* __restrict__ Q,   // [B,NH,S,HSZ]
                const __bf16* __restrict__ K,   // [B,NH,S,HSZ]
                const __bf16* __restrict__ Vt,  // [B,NH,HSZ,S]
                const int* __restrict__ pmask,  // [B,S]
                __bf16* __restrict__ O) {       // [MTOT][HDIM]
  __shared__ __align__(64) __bf16 ptile[8][16][32];
  const int lane = threadIdx.x & 31;
  const int half = lane >> 4;
  const int l15  = lane & 15;
  const int wv   = threadIdx.x >> 5;
  const int wid  = blockIdx.x * 8 + wv;
  const int q0   = (wid & 127) << 4;   // SEQ/16 = 128 query tiles
  const int bh   = wid >> 7;
  const int b    = bh >> 4;
  const int h    = bh & 15;

  const __bf16* Qh = Q + (size_t)bh * SEQ * HSZ;
  const __bf16* Kh = K + (size_t)bh * SEQ * HSZ;
  const __bf16* Vh = Vt + (size_t)bh * HSZ * SEQ;
  const int* pm = pmask + b * SEQ;

  v16bf qf[4];
#pragma unroll
  for (int c = 0; c < 4; ++c) qf[c] = load_a_frag(Qh, HSZ, q0, c * 32, l15, half);

  v8f acc[8] = {};
  float rm[8], rs[8];
#pragma unroll
  for (int r = 0; r < 8; ++r) { rm[r] = -1e30f; rs[r] = 0.f; }

  const int kend = ((q0 + 16 + 31) >> 5) << 5;   // causal bound rounded to 32
  for (int kt = 0; kt < kend; kt += 32) {
    // scores: Q(16xHSZ) * K^T -> two 16x16 fp32 fragments
    v8f s0 = {}, s1 = {};
#pragma unroll
    for (int c = 0; c < 4; ++c) {
      v16bf b0 = load_b_frag(Kh, HSZ, kt, c * 32, l15, half);
      s0 = wmma_bf16(qf[c], b0, s0);
      v16bf b1 = load_b_frag(Kh, HSZ, kt + 16, c * 32, l15, half);
      s1 = wmma_bf16(qf[c], b1, s1);
    }
    const float pad0 = pm[kt + l15] ? 0.f : -10000.f;
    const float pad1 = pm[kt + 16 + l15] ? 0.f : -10000.f;
#pragma unroll
    for (int r = 0; r < 8; ++r) {
      const int row = q0 + r + half * 8;
      float x0 = s0[r] * SM_SCALE + pad0 + ((kt + l15) > row ? -1e9f : 0.f);
      float x1 = s1[r] * SM_SCALE + pad1 + ((kt + 16 + l15) > row ? -1e9f : 0.f);
      const float tmax = red16_max(fmaxf(x0, x1));
      const float mn   = fmaxf(rm[r], tmax);
      const float corr = __expf(rm[r] - mn);
      const float p0   = __expf(x0 - mn);
      const float p1   = __expf(x1 - mn);
      rs[r] = rs[r] * corr + red16_sum(p0 + p1);
      rm[r] = mn;
      ptile[wv][r + half * 8][l15]      = (__bf16)p0;
      ptile[wv][r + half * 8][16 + l15] = (__bf16)p1;
#pragma unroll
      for (int f = 0; f < 8; ++f) acc[f][r] *= corr;
    }
    // same-wave LDS RAW: DS ops are in-order per wave; wait out DScnt
    asm volatile("s_wait_dscnt 0" ::: "memory");
    v16bf pA = load_a_frag(&ptile[wv][0][0], 32, 0, 0, l15, half);
#pragma unroll
    for (int f = 0; f < 8; ++f) {
      v16bf bv = load_b_frag(Vh, SEQ, f * 16, kt, l15, half);  // Vt: [hd][s]
      acc[f] = wmma_bf16(pA, bv, acc[f]);
    }
  }

  const int mbase = b * SEQ + q0;
  float rinv[8];
#pragma unroll
  for (int r = 0; r < 8; ++r) rinv[r] = 1.0f / rs[r];
#pragma unroll
  for (int f = 0; f < 8; ++f) {
#pragma unroll
    for (int r = 0; r < 8; ++r) {
      const float o = acc[f][r] * rinv[r];
      const int m = mbase + r + half * 8;
      O[(size_t)m * HDIM + h * HSZ + f * 16 + l15] = (__bf16)o;
    }
  }
}

// ---------------------------------------------------------------------------
extern "C" void kernel_launch(void* const* d_in, const int* in_sizes, int n_in,
                              void* d_out, int out_size, void* d_ws, size_t ws_size,
                              hipStream_t stream) {
  const float* x    = (const float*)d_in[0];
  const float* fcos = (const float*)d_in[1];
  const float* fsin = (const float*)d_in[2];
  const int*   pmsk = (const int*)d_in[3];
  const float* Wq   = (const float*)d_in[4];
  const float* Wk   = (const float*)d_in[5];
  const float* Wv   = (const float*)d_in[6];
  const float* Wo   = (const float*)d_in[7];
  float* out = (float*)d_out;

  char* ws = (char*)d_ws;
  const size_t MB = (size_t)1 << 20;
  __bf16* Xb  = (__bf16*)(ws +  0 * MB);   // [MTOT][HDIM]        16 MB
  __bf16* Wqb = (__bf16*)(ws + 16 * MB);   // [HDIM][HDIM]         8 MB
  __bf16* Wkb = (__bf16*)(ws + 24 * MB);
  __bf16* Wvb = (__bf16*)(ws + 32 * MB);
  __bf16* Wob = (__bf16*)(ws + 40 * MB);
  __bf16* Qb  = (__bf16*)(ws + 48 * MB);   // [B,NH,S,HSZ]        16 MB
  __bf16* Kb  = (__bf16*)(ws + 64 * MB);
  __bf16* Vtb = (__bf16*)(ws + 80 * MB);   // [B,NH,HSZ,S]
  __bf16* Ob  = (__bf16*)(ws + 96 * MB);   // [MTOT][HDIM]

  const int NX = MTOT * HDIM;   // 8M elems
  const int NW = HDIM * HDIM;   // 4M elems
  cast_f32_bf16<<<2048, 256, 0, stream>>>(x,  Xb,  NX);
  cast_f32_bf16<<<2048, 256, 0, stream>>>(Wq, Wqb, NW);
  cast_f32_bf16<<<2048, 256, 0, stream>>>(Wk, Wkb, NW);
  cast_f32_bf16<<<2048, 256, 0, stream>>>(Wv, Wvb, NW);
  cast_f32_bf16<<<2048, 256, 0, stream>>>(Wo, Wob, NW);

  const int gemm_blocks = (MTOT / 16) * (HDIM / 128) / 8;  // 512 blocks x 8 waves
  gemm16x128<MODE_ROPE><<<gemm_blocks, 256, 0, stream>>>(Xb, Wqb, Qb, nullptr, fcos, fsin);
  gemm16x128<MODE_ROPE><<<gemm_blocks, 256, 0, stream>>>(Xb, Wkb, Kb, nullptr, fcos, fsin);
  gemm16x128<MODE_V><<<gemm_blocks, 256, 0, stream>>>(Xb, Wvb, Vtb, nullptr, nullptr, nullptr);

  const int attn_blocks = (BATCH * NHEADS * (SEQ / 16)) / 8;  // 512
  flash_attn<<<attn_blocks, 256, 0, stream>>>(Qb, Kb, Vtb, pmsk, Ob);

  gemm16x128<MODE_OUT><<<gemm_blocks, 256, 0, stream>>>(Ob, Wob, nullptr, out, nullptr, nullptr);
}
